// MultiHeadAttention_64905545777763
// MI455X (gfx1250) — matrix operand order repaired and checked by reading, not compiled, not run on hardware
//
#include <hip/hip_runtime.h>
#include <hip/hip_bf16.h>

// ---------------------------------------------------------------------------
// MI455X (gfx1250) transformer cross-attention block.
// - All matmuls on v_wmma_f32_16x16x32_f16 (f16 in, fp32 accumulate).
// - LDS tiles staged with GLOBAL_LOAD_ASYNC_TO_LDS_B128 (ASYNCcnt) when the
//   toolchain exposes the builtin; double-buffered so the copy overlaps WMMA.
// - V is produced pre-transposed by the V-projection GEMM so P.V fragments are
//   contiguous global loads (no per-tile LDS transpose).
// - Lengths are multiples of 128 (setup_inputs) => packed-coordinate tiling
//   needs no masks. Softmax/LayerNorm math stays fp32.
// ---------------------------------------------------------------------------

#define D_MODEL 512
#define D_FF    2048
#define DHEAD   64

typedef __attribute__((ext_vector_type(16))) _Float16 v16h;
typedef __attribute__((ext_vector_type(8)))  _Float16 v8h;
typedef __attribute__((ext_vector_type(4)))  _Float16 v4h;
typedef __attribute__((ext_vector_type(8)))  float    v8f;
typedef __attribute__((ext_vector_type(4)))  float    v4f;
typedef __attribute__((ext_vector_type(4)))  int      v4i;

#define AS1 __attribute__((address_space(1)))
#define AS3 __attribute__((address_space(3)))

#if defined(__has_builtin)
#  if __has_builtin(__builtin_amdgcn_global_load_async_to_lds_b128)
#    define HAS_ASYNC_COPY 1
#  endif
#endif
#ifndef HAS_ASYNC_COPY
#  define HAS_ASYNC_COPY 0
#endif

// 16-byte global -> LDS copy. Async (ASYNCcnt) when available.
__device__ __forceinline__ void async_copy16(const _Float16* g, _Float16* l) {
#if HAS_ASYNC_COPY
  __builtin_amdgcn_global_load_async_to_lds_b128((AS1 v4i*)(size_t)g,
                                                 (AS3 v4i*)l, 0, 0);
#else
  *(v8h*)l = *(const v8h*)g;
#endif
}

__device__ __forceinline__ void wait_async0() {
#if HAS_ASYNC_COPY
#  if __has_builtin(__builtin_amdgcn_s_wait_asynccnt)
  __builtin_amdgcn_s_wait_asynccnt(0);
#  else
  asm volatile("s_wait_asynccnt 0x0" ::: "memory");
#  endif
#endif
}

__device__ __forceinline__ v8f wmma_f16(v16h a, v16h b, v8f c) {
  // (neg_a, A, neg_b, B, c_mod, C, reuse_a, reuse_b)
  return __builtin_amdgcn_wmma_f32_16x16x32_f16(false, a, false, b, (short)0, c,
                                                false, false);
}

// A fragment (16M x 32K), row-major source. base = &A[row0][k0], ld elems.
// Layout: lanes 0-15 row M=lane hold K 0..15 contiguous; lanes 16-31 hold
// K 16..31.
__device__ __forceinline__ v16h frag_a(const _Float16* base, int ld, int lane) {
  int m  = lane & 15;
  int kb = (lane >> 4) * 16;
  const _Float16* p = base + (size_t)m * ld + kb;
  v8h lo = *(const v8h*)(p);
  v8h hi = *(const v8h*)(p + 8);
  v16h r;
#pragma unroll
  for (int i = 0; i < 8; ++i) { r[i] = lo[i]; r[i + 8] = hi[i]; }
  return r;
}

// B fragment (32K x 16N) from N-major (transposed) storage Bt[n][k].
// Lanes 0-15: column N=lane, K 0..15 contiguous; lanes 16-31: K 16..31.
__device__ __forceinline__ v16h frag_b(const _Float16* base, int ld, int lane) {
  int n  = lane & 15;
  int kb = (lane >> 4) * 16;
  return *(const v16h*)(base + (size_t)n * ld + kb);
}

// ---------------------------------------------------------------------------
__global__ __launch_bounds__(256) void cvt_f16_kernel(
    const float* __restrict__ in, _Float16* __restrict__ out, int n4) {
  int i = blockIdx.x * 256 + threadIdx.x;
  if (i >= n4) return;
  v4f v = *(const v4f*)(in + (size_t)i * 4);
  v4h o;
#pragma unroll
  for (int k = 0; k < 4; ++k) o[k] = (_Float16)v[k];
  *(v4h*)(out + (size_t)i * 4) = o;
}

// in[K][N] (f32, row-major) -> out[N][K] (f16, N-major). K,N multiples of 32.
__global__ __launch_bounds__(256) void transpose_cvt_kernel(
    const float* __restrict__ in, _Float16* __restrict__ out, int K, int N) {
  __shared__ _Float16 tile[32][33];
  int n0 = blockIdx.x * 32, k0 = blockIdx.y * 32;
  int tx = threadIdx.x & 31, ty = threadIdx.x >> 5;  // 32x8
#pragma unroll
  for (int i = 0; i < 32; i += 8)
    tile[ty + i][tx] = (_Float16)in[(size_t)(k0 + ty + i) * N + n0 + tx];
  __syncthreads();
#pragma unroll
  for (int i = 0; i < 32; i += 8)
    out[(size_t)(n0 + ty + i) * K + k0 + tx] = tile[tx][ty + i];
}

// ---------------------------------------------------------------------------
// C[M,N] = act(scale * A[M,K] x Bt[N,K]^T + bias). 8 waves: block tile 128x64,
// each wave one 16-row strip x 64 cols. B tile double-buffered in LDS via
// async copies so the next tile's fetch overlaps the current WMMAs.
// storeT=1 stores the f16 result column-major (Ch[col*M + row]).
__global__ __launch_bounds__(256) void gemm_f16_kernel(
    const _Float16* __restrict__ A, const _Float16* __restrict__ Bt,
    const float* __restrict__ bias, float* __restrict__ Cf,
    _Float16* __restrict__ Ch, int M, int N, int K, float scale, int relu,
    int storeT) {
  __shared__ __align__(64) _Float16 sB[2][64][32];
  int tid = threadIdx.x, lane = tid & 31, wave = tid >> 5;
  int m0 = blockIdx.x * 128 + wave * 16;
  int n0 = blockIdx.y * 64;
  int br = tid >> 2, bc = (tid & 3) * 8;  // 64 rows x 32 k, 16B per thread
  const _Float16* bsrc = Bt + (size_t)(n0 + br) * K + bc;
  v8f acc[4] = {};
  async_copy16(bsrc, &sB[0][br][bc]);
  wait_async0();
  __syncthreads();
  int cur = 0;
  for (int kk = 0; kk < K; kk += 32) {
    if (kk + 32 < K) async_copy16(bsrc + kk + 32, &sB[cur ^ 1][br][bc]);
    v16h a = frag_a(A + (size_t)m0 * K + kk, K, lane);
#pragma unroll
    for (int j = 0; j < 4; ++j)
      acc[j] = wmma_f16(a, frag_b(&sB[cur][j * 16][0], 32, lane), acc[j]);
    wait_async0();
    __syncthreads();
    cur ^= 1;
  }
  int half = lane >> 4, nn = lane & 15;
#pragma unroll
  for (int j = 0; j < 4; ++j) {
    int col  = n0 + j * 16 + nn;
    float bv = bias ? bias[col] : 0.0f;
#pragma unroll
    for (int r = 0; r < 8; ++r) {
      int row = m0 + half * 8 + r;  // C layout: lanes<16 -> M=r, else M=8+r
      float v = acc[j][r] * scale + bv;
      if (relu) v = fmaxf(v, 0.0f);
      if (Cf) Cf[(size_t)row * N + col] = v;
      if (Ch) {
        if (storeT) Ch[(size_t)col * M + row] = (_Float16)v;
        else        Ch[(size_t)row * N + col] = (_Float16)v;
      }
    }
  }
}

// ---------------------------------------------------------------------------
// Flash cross-attention. Grid: (sum(len_x)/128, NHEAD). 8 waves, each owns 16
// query rows. K tiles (32 keys) double-buffered in LDS via async copies.
// V is pre-transposed global ([D_MODEL][total]) so P.V B-fragments are
// contiguous global loads. 1/sqrt(dh) pre-folded into Q.
__global__ __launch_bounds__(256) void flash_attn_kernel(
    const _Float16* __restrict__ Q, const _Float16* __restrict__ Kv,
    const _Float16* __restrict__ Vt, _Float16* __restrict__ O,
    const long long* __restrict__ lx, const long long* __restrict__ lm,
    int nb, int total) {
  __shared__ __align__(64) _Float16 sK[2][32][64];   // 32 keys x 64 dh, x2
  __shared__ __align__(64) _Float16 sP[8][16][32];   // per-wave P staging
  int tid = threadIdx.x, lane = tid & 31, wave = tid >> 5;
  int h = blockIdx.y;
  int qt = blockIdx.x;
  long long rsx = 0, rsm = 0;
  int b = 0;
  for (; b < nb; ++b) {           // map flat q-tile -> (batch, tile in batch)
    int nt = (int)(lx[b] >> 7);   // lengths are multiples of 128
    if (qt < nt) break;
    qt -= nt; rsx += lx[b]; rsm += lm[b];
  }
  int L = (int)lm[b];
  size_t q0 = (size_t)rsx + (size_t)qt * 128 + (size_t)wave * 16;
  const _Float16* Qb = Q + q0 * D_MODEL + h * DHEAD;
  const _Float16* Kb = Kv + (size_t)rsm * D_MODEL + h * DHEAD;
  const _Float16* Vb = Vt + (size_t)h * DHEAD * total + rsm;  // V^T rows
  int key = tid >> 3, kc = (tid & 7) * 8;  // cooperative 32x64 staging
  const _Float16* ksrc = Kb + (size_t)key * D_MODEL + kc;
  async_copy16(ksrc, &sK[0][key][kc]);
  v16h qa0 = frag_a(Qb, D_MODEL, lane);       // dh 0..31
  v16h qa1 = frag_a(Qb + 32, D_MODEL, lane);  // dh 32..63
  float mr[8], lr[8];
  v8f acc[4] = {};
#pragma unroll
  for (int r = 0; r < 8; ++r) { mr[r] = -1e30f; lr[r] = 0.0f; }
  int half = lane >> 4, nn = lane & 15;
  wait_async0();
  __syncthreads();
  int cur = 0;
  for (int kt = 0; kt < L; kt += 32) {
    if (kt + 32 < L)
      async_copy16(ksrc + (size_t)(kt + 32) * D_MODEL, &sK[cur ^ 1][key][kc]);
    // S[16q x 32k] = Q . K^T  (K rows act as N-major B for the WMMA)
    v8f s0 = {}, s1 = {};
    s0 = wmma_f16(qa0, frag_b(&sK[cur][0][0], 64, lane), s0);
    s0 = wmma_f16(qa1, frag_b(&sK[cur][0][32], 64, lane), s0);
    s1 = wmma_f16(qa0, frag_b(&sK[cur][16][0], 64, lane), s1);
    s1 = wmma_f16(qa1, frag_b(&sK[cur][16][32], 64, lane), s1);
    // online softmax; row = 8*half + r lives in one 16-lane group
#pragma unroll
    for (int r = 0; r < 8; ++r) {
      float mx = fmaxf(s0[r], s1[r]);
#pragma unroll
      for (int off = 8; off > 0; off >>= 1)
        mx = fmaxf(mx, __shfl_xor(mx, off, 16));
      float mnew = fmaxf(mr[r], mx);
      float corr = __expf(mr[r] - mnew);
      float e0 = __expf(s0[r] - mnew);
      float e1 = __expf(s1[r] - mnew);
      float rs = e0 + e1;
#pragma unroll
      for (int off = 8; off > 0; off >>= 1) rs += __shfl_xor(rs, off, 16);
      lr[r] = lr[r] * corr + rs;
      mr[r] = mnew;
#pragma unroll
      for (int j = 0; j < 4; ++j) acc[j][r] = acc[j][r] * corr;
      int row = half * 8 + r;
      sP[wave][row][nn] = (_Float16)e0;
      sP[wave][row][nn + 16] = (_Float16)e1;
    }
    // O += P . V  (P via wave-private LDS reshape; V^T frags from global)
    v16h pa = frag_a(&sP[wave][0][0], 32, lane);
#pragma unroll
    for (int j = 0; j < 4; ++j)
      acc[j] = wmma_f16(
          pa, frag_b(Vb + (size_t)(j * 16) * total + kt, total, lane), acc[j]);
    wait_async0();
    __syncthreads();
    cur ^= 1;
  }
#pragma unroll
  for (int j = 0; j < 4; ++j) {
    int col = h * DHEAD + j * 16 + nn;
#pragma unroll
    for (int r = 0; r < 8; ++r) {
      size_t row = q0 + (size_t)(half * 8 + r);
      O[row * D_MODEL + col] = (_Float16)(acc[j][r] / lr[r]);
    }
  }
}

// ---------------------------------------------------------------------------
// out = LayerNorm(X + Y) * g + b. One wave per row (16 elems/lane).
__global__ __launch_bounds__(256) void add_ln_kernel(
    const float* __restrict__ X, const float* __restrict__ Y,
    const float* __restrict__ g, const float* __restrict__ bta,
    float* __restrict__ o32, _Float16* __restrict__ o16) {
  int row = blockIdx.x * 8 + (threadIdx.x >> 5);
  int lane = threadIdx.x & 31;
  const float* xr = X + (size_t)row * D_MODEL;
  const float* yr = Y + (size_t)row * D_MODEL;
  float v[16];
  float sum = 0.0f;
#pragma unroll
  for (int i = 0; i < 16; ++i) {
    int c = lane + 32 * i;
    v[i] = xr[c] + yr[c];
    sum += v[i];
  }
#pragma unroll
  for (int off = 16; off > 0; off >>= 1) sum += __shfl_xor(sum, off, 32);
  float mu = sum * (1.0f / D_MODEL);
  float var = 0.0f;
#pragma unroll
  for (int i = 0; i < 16; ++i) { float d = v[i] - mu; var += d * d; }
#pragma unroll
  for (int off = 16; off > 0; off >>= 1) var += __shfl_xor(var, off, 32);
  float rstd = rsqrtf(var * (1.0f / D_MODEL) + 1e-6f);
#pragma unroll
  for (int i = 0; i < 16; ++i) {
    int c = lane + 32 * i;
    float o = (v[i] - mu) * rstd * g[c] + bta[c];
    if (o32) o32[(size_t)row * D_MODEL + c] = o;
    if (o16) o16[(size_t)row * D_MODEL + c] = (_Float16)o;
  }
}

// ---------------------------------------------------------------------------
extern "C" void kernel_launch(void* const* d_in, const int* in_sizes, int n_in,
                              void* d_out, int out_size, void* d_ws,
                              size_t ws_size, hipStream_t stream) {
  const float* x = (const float*)d_in[0];
  const float* mem = (const float*)d_in[1];
  const long long* lx = (const long long*)d_in[2];
  const long long* lm = (const long long*)d_in[3];
  const float* Wq = (const float*)d_in[4];
  const float* Wk = (const float*)d_in[5];
  const float* Wv = (const float*)d_in[6];
  const float* Wo = (const float*)d_in[7];
  const float* ln1s = (const float*)d_in[8];
  const float* ln1b = (const float*)d_in[9];
  const float* W1 = (const float*)d_in[10];
  const float* b1 = (const float*)d_in[11];
  const float* W2 = (const float*)d_in[12];
  const float* b2 = (const float*)d_in[13];
  const float* ln2s = (const float*)d_in[14];
  const float* ln2b = (const float*)d_in[15];
  const int nb = in_sizes[2];
  const int total = in_sizes[0] / D_MODEL;  // 4608 packed tokens

  size_t off = 0;
  auto alloc = [&](size_t bytes) -> char* {
    char* p = (char*)d_ws + off;
    off += (bytes + 255) & ~(size_t)255;
    return p;
  };
  const size_t tokh = (size_t)total * D_MODEL;  // token-matrix elems
  _Float16* x16 = (_Float16*)alloc(tokh * 2);
  _Float16* mem16 = (_Float16*)alloc(tokh * 2);
  _Float16* WqT = (_Float16*)alloc((size_t)D_MODEL * D_MODEL * 2);
  _Float16* WkT = (_Float16*)alloc((size_t)D_MODEL * D_MODEL * 2);
  _Float16* WvT = (_Float16*)alloc((size_t)D_MODEL * D_MODEL * 2);
  _Float16* WoT = (_Float16*)alloc((size_t)D_MODEL * D_MODEL * 2);
  _Float16* W1T = (_Float16*)alloc((size_t)D_MODEL * D_FF * 2);
  _Float16* W2T = (_Float16*)alloc((size_t)D_FF * D_MODEL * 2);
  _Float16* q16 = (_Float16*)alloc(tokh * 2);
  _Float16* k16 = (_Float16*)alloc(tokh * 2);
  _Float16* vt16 = (_Float16*)alloc(tokh * 2);  // V^T: [D_MODEL][total]
  _Float16* attn16 = (_Float16*)alloc(tokh * 2);
  float* proj32 = (float*)alloc(tokh * 4);
  float* h32 = (float*)alloc(tokh * 4);
  _Float16* h16 = (_Float16*)alloc(tokh * 2);
  _Float16* ff1 = (_Float16*)alloc((size_t)total * D_FF * 2);
  float* ff2 = (float*)alloc(tokh * 4);

  const int n4 = (int)(tokh / 4);
  cvt_f16_kernel<<<(n4 + 255) / 256, 256, 0, stream>>>(x, x16, n4);
  cvt_f16_kernel<<<(n4 + 255) / 256, 256, 0, stream>>>(mem, mem16, n4);

  transpose_cvt_kernel<<<dim3(D_MODEL / 32, D_MODEL / 32), 256, 0, stream>>>(
      Wq, WqT, D_MODEL, D_MODEL);
  transpose_cvt_kernel<<<dim3(D_MODEL / 32, D_MODEL / 32), 256, 0, stream>>>(
      Wk, WkT, D_MODEL, D_MODEL);
  transpose_cvt_kernel<<<dim3(D_MODEL / 32, D_MODEL / 32), 256, 0, stream>>>(
      Wv, WvT, D_MODEL, D_MODEL);
  transpose_cvt_kernel<<<dim3(D_MODEL / 32, D_MODEL / 32), 256, 0, stream>>>(
      Wo, WoT, D_MODEL, D_MODEL);
  transpose_cvt_kernel<<<dim3(D_FF / 32, D_MODEL / 32), 256, 0, stream>>>(
      W1, W1T, D_MODEL, D_FF);
  transpose_cvt_kernel<<<dim3(D_MODEL / 32, D_FF / 32), 256, 0, stream>>>(
      W2, W2T, D_FF, D_MODEL);

  const float qscale = 0.125f;  // 1/sqrt(64) folded into Q
  dim3 gproj(total / 128, D_MODEL / 64);
  gemm_f16_kernel<<<gproj, 256, 0, stream>>>(x16, WqT, nullptr, nullptr, q16,
                                             total, D_MODEL, D_MODEL, qscale,
                                             0, 0);
  gemm_f16_kernel<<<gproj, 256, 0, stream>>>(mem16, WkT, nullptr, nullptr, k16,
                                             total, D_MODEL, D_MODEL, 1.0f, 0,
                                             0);
  gemm_f16_kernel<<<gproj, 256, 0, stream>>>(mem16, WvT, nullptr, nullptr,
                                             vt16, total, D_MODEL, D_MODEL,
                                             1.0f, 0, /*storeT=*/1);

  // sum(len_x) == total and each length is a multiple of 128
  flash_attn_kernel<<<dim3(total / 128, 8), 256, 0, stream>>>(
      q16, k16, vt16, attn16, lx, lm, nb, total);

  gemm_f16_kernel<<<gproj, 256, 0, stream>>>(attn16, WoT, nullptr, proj32,
                                             nullptr, total, D_MODEL, D_MODEL,
                                             1.0f, 0, 0);
  add_ln_kernel<<<total / 8, 256, 0, stream>>>(x, proj32, ln1s, ln1b, h32, h16);

  dim3 gff1(total / 128, D_FF / 64);
  gemm_f16_kernel<<<gff1, 256, 0, stream>>>(h16, W1T, b1, nullptr, ff1, total,
                                            D_FF, D_MODEL, 1.0f, 1, 0);
  gemm_f16_kernel<<<gproj, 256, 0, stream>>>(ff1, W2T, b2, ff2, nullptr, total,
                                             D_MODEL, D_FF, 1.0f, 0, 0);
  add_ln_kernel<<<total / 8, 256, 0, stream>>>(h32, ff2, ln2s, ln2b,
                                               (float*)d_out, nullptr);
}
